// Non_Local_Network_83837761618153
// MI455X (gfx1250) — matrix operand order repaired and checked by reading, not compiled, not run on hardware
//
#include <hip/hip_runtime.h>
#include <hip/hip_bf16.h>
#include <stdint.h>

#define NB 4
#define NC 192
#define NH 64
#define NW 64
#define NHW 4096
#define NKW 1728          // NC * 9
#define NPW 66            // padded width
#define NPHW 4356         // 66*66

typedef __attribute__((ext_vector_type(16))) _Float16 v16h;
typedef __attribute__((ext_vector_type(8)))  float    v8f;
typedef __attribute__((ext_vector_type(4)))  uint32_t u32x4;

union Frag { v16h h; u32x4 q[2]; };
union Acc  { v8f  v; float f[8]; };

__device__ __forceinline__ unsigned short f2h(float x) {
    union { _Float16 h; unsigned short u; } c; c.h = (_Float16)x; return c.u;
}

__device__ __forceinline__ v8f wmma16(const Frag& a, const Frag& b, const v8f& c) {
    return __builtin_amdgcn_wmma_f32_16x16x32_f16(false, a.h, false, b.h,
                                                  (short)0, c, false, false);
}

// A-fragment (16x32 f16): lane holds contiguous halfs [g*8, g*8+8) and [16+g*8, ...)
__device__ __forceinline__ void loadA(Frag& f, const unsigned short* p, int g) {
    f.q[0] = *(const u32x4*)(p + g * 8);
    f.q[1] = *(const u32x4*)(p + 16 + g * 8);
}
// B-fragment (32x16 f16): lane holds contiguous halfs [g*8, g*8+8) and [16+g*8, ...)
__device__ __forceinline__ void loadB(Frag& f, const unsigned short* p, int g) {
    f.q[0] = *(const u32x4*)(p + g * 8);
    f.q[1] = *(const u32x4*)(p + 16 + g * 8);
}

// ---------------- prep kernels -----------------------------------------
__global__ void fill_zero_u16(unsigned short* __restrict__ dst, int n) {
    for (int i = blockIdx.x * blockDim.x + threadIdx.x; i < n; i += gridDim.x * blockDim.x)
        dst[i] = 0;
}

__global__ void cvt_f16(const float* __restrict__ src, unsigned short* __restrict__ dst, int n) {
    for (int i = blockIdx.x * blockDim.x + threadIdx.x; i < n; i += gridDim.x * blockDim.x)
        dst[i] = f2h(src[i]);
}

// fp32 [B,C,64,64] -> f16 zero-padded HWC [B][66*66][C]
__global__ void pack_pad_hwc(const float* __restrict__ src, unsigned short* __restrict__ dst,
                             int n) {
    for (int i = blockIdx.x * blockDim.x + threadIdx.x; i < n; i += gridDim.x * blockDim.x) {
        int c = i % NC;
        int p = (i / NC) % NPHW;
        int b = i / (NC * NPHW);
        int py = p / NPW, px = p % NPW;
        unsigned short h = 0;
        if (py >= 1 && py <= NH && px >= 1 && px <= NW)
            h = f2h(src[((size_t)b * NC + c) * NHW + (py - 1) * NW + (px - 1)]);
        dst[i] = h;
    }
}

// fp32 [O,C,3,3] -> f16 [O][9][C]
__global__ void pack_wgt(const float* __restrict__ src, unsigned short* __restrict__ dst, int n) {
    for (int i = blockIdx.x * blockDim.x + threadIdx.x; i < n; i += gridDim.x * blockDim.x) {
        int c = i % NC;
        int tap = (i / NC) % 9;
        int o = i / (NC * 9);
        dst[i] = f2h(src[((size_t)o * NC + c) * 9 + tap]);
    }
}

// f16 [B,C,64,64] view of a fp32 [B,C,h,w] nearest-upsample -> CHW f16 [B][C][4096]
__global__ void upsample_f16(const float* __restrict__ src, unsigned short* __restrict__ dst,
                             int f, int n) {
    int sw = NW / f;
    for (int i = blockIdx.x * blockDim.x + threadIdx.x; i < n; i += gridDim.x * blockDim.x) {
        int x = i & 63, y = (i >> 6) & 63, bc = i >> 12;
        dst[i] = f2h(src[(size_t)bc * sw * sw + (y / f) * sw + (x / f)]);
    }
}

// ---------------- conv3x3: tap-major implicit GEMM ----------------------
// xin  f16 padded HWC [B][66*66][C]
// wgt  f16 [O][9][C]
// mode 0: write padded HWC (for the conv1->conv2 chain)
// mode 1: write plain HWC [B][4096][C] with fp32 residual added
__global__ void __launch_bounds__(128)
conv16(const unsigned short* __restrict__ xin, const unsigned short* __restrict__ wgt,
       const float* __restrict__ bias, const float* __restrict__ resid,
       unsigned short* __restrict__ outp, int mode) {
    const int lane = threadIdx.x & 31;
    const int g = lane >> 4, nn = lane & 15;
    int wave = blockIdx.x * 4 + (threadIdx.x >> 5);
    int b   = wave / (NH * 48);
    int rem = wave % (NH * 48);
    int y   = rem / 48;
    int t   = rem % 48;
    int c0  = (t >> 2) * 16;
    int x0  = (t & 3) * 16;

    Acc acc;
#pragma unroll
    for (int r = 0; r < 8; r++) acc.f[r] = 0.f;

    const unsigned short* xb = xin + (size_t)b * NPHW * NC;
    const unsigned short* wb = wgt + (size_t)(c0 + nn) * NKW;

#pragma unroll
    for (int tap = 0; tap < 9; tap++) {
        const int dy = tap / 3 - 1, dx = tap % 3 - 1;
        const unsigned short* arow = xb + (size_t)((y + dy + 1) * NPW + x0 + nn + dx + 1) * NC;
        const unsigned short* wrow = wb + tap * NC;
#pragma unroll
        for (int kk = 0; kk < 6; kk++) {
            Frag af, bf;
            loadA(af, arow + kk * 32, g);
            loadB(bf, wrow + kk * 32, g);
            acc.v = wmma16(af, bf, acc.v);
        }
    }

    int cc = c0 + nn;
    float bsv = bias[cc];
#pragma unroll
    for (int r = 0; r < 8; r++) {
        int M = r + 8 * g;
        int x = x0 + M;
        float val = acc.f[r] + bsv;
        val = val > 0.f ? val : 0.01f * val;           // LeakyReLU(0.01)
        if (mode == 0) {
            outp[((size_t)b * NPHW + (y + 1) * NPW + x + 1) * NC + cc] = f2h(val);
        } else {
            val += resid[(((size_t)b * NC + cc) * NH + y) * NW + x];
            outp[((size_t)b * NHW + y * NW + x) * NC + cc] = f2h(val);
        }
    }
}

// ---------------- attention pass 1: per-row softmax stats ---------------
__global__ void __launch_bounds__(128)
attn_stats(const unsigned short* __restrict__ f1, const unsigned short* __restrict__ f2t,
           float* __restrict__ mout, float* __restrict__ lout) {
    const int lane = threadIdx.x & 31;
    const int g = lane >> 4, nn = lane & 15;
    int wave = blockIdx.x * 4 + (threadIdx.x >> 5);
    int b = wave >> 8;
    int q0 = (wave & 255) * 16;

    Frag qf[6];
#pragma unroll
    for (int kk = 0; kk < 6; kk++)
        loadA(qf[kk], f1 + ((size_t)b * NHW + q0 + nn) * NC + kk * 32, g);

    float mrun[8], lrun[8];
#pragma unroll
    for (int r = 0; r < 8; r++) { mrun[r] = -1e30f; lrun[r] = 0.f; }

    const unsigned short* kbase = f2t + (size_t)b * NHW * NC;
#pragma unroll 1
    for (int jb = 0; jb < NHW; jb += 32) {
        __builtin_prefetch(kbase + (size_t)(jb + 32 + nn) * NC, 0, 1);
        Acc slo, shi;
#pragma unroll
        for (int r = 0; r < 8; r++) { slo.f[r] = 0.f; shi.f[r] = 0.f; }
#pragma unroll
        for (int kk = 0; kk < 6; kk++) {
            Frag bl, bh;
            loadB(bl, kbase + (size_t)(jb + nn) * NC + kk * 32, g);
            loadB(bh, kbase + (size_t)(jb + 16 + nn) * NC + kk * 32, g);
            slo.v = wmma16(qf[kk], bl, slo.v);
            shi.v = wmma16(qf[kk], bh, shi.v);
        }
#pragma unroll
        for (int r = 0; r < 8; r++) {
            float tm = fmaxf(slo.f[r], shi.f[r]);
            tm = fmaxf(tm, __shfl_xor(tm, 1, 32));
            tm = fmaxf(tm, __shfl_xor(tm, 2, 32));
            tm = fmaxf(tm, __shfl_xor(tm, 4, 32));
            tm = fmaxf(tm, __shfl_xor(tm, 8, 32));
            float mnew = fmaxf(mrun[r], tm);
            float ps = __expf(slo.f[r] - mnew) + __expf(shi.f[r] - mnew);
            ps += __shfl_xor(ps, 1, 32);
            ps += __shfl_xor(ps, 2, 32);
            ps += __shfl_xor(ps, 4, 32);
            ps += __shfl_xor(ps, 8, 32);
            lrun[r] = lrun[r] * __expf(mrun[r] - mnew) + ps;
            mrun[r] = mnew;
        }
    }
    if (nn == 0) {
#pragma unroll
        for (int r = 0; r < 8; r++) {
            int i = q0 + r + 8 * g;
            mout[(size_t)b * NHW + i] = mrun[r];
            lout[(size_t)b * NHW + i] = lrun[r];
        }
    }
}

// ---------------- attention pass 2: P·V with fused query pooling --------
__device__ __forceinline__ int qmap(int mode, int t, int M) {
    if (mode == 0) return t * 16 + M;
    if (mode == 1) { int y0 = (t >> 3) * 2, x0 = (t & 7) * 8;
                     return (y0 + (M >> 3)) * NW + x0 + (M & 7); }
    int y0 = (t >> 4) * 4, x0 = (t & 15) * 4;
    return (y0 + (M >> 2)) * NW + x0 + (M & 3);
}

__global__ void __launch_bounds__(128)
attn_out(const unsigned short* __restrict__ f1, const unsigned short* __restrict__ f2t,
         const unsigned short* __restrict__ vt, const float* __restrict__ mbuf,
         const float* __restrict__ lbuf, float* __restrict__ outp, int mode) {
    __shared__ unsigned short ptile[4][512];          // per-wave 16x32 f16 P tile
    const int lane = threadIdx.x & 31;
    const int g = lane >> 4, nn = lane & 15;
    const int wv = threadIdx.x >> 5;
    int wave = blockIdx.x * 4 + wv;
    int b = wave >> 8;
    int t = wave & 255;

    Frag qf[6];
#pragma unroll
    for (int kk = 0; kk < 6; kk++)
        loadA(qf[kk], f1 + ((size_t)b * NHW + qmap(mode, t, nn)) * NC + kk * 32, g);

    float mrow[8], linv[8];
#pragma unroll
    for (int r = 0; r < 8; r++) {
        int i = qmap(mode, t, r + 8 * g);
        mrow[r] = mbuf[(size_t)b * NHW + i];
        linv[r] = 1.f / lbuf[(size_t)b * NHW + i];
    }

    Acc acc[12];
#pragma unroll
    for (int cg = 0; cg < 12; cg++)
#pragma unroll
        for (int r = 0; r < 8; r++) acc[cg].f[r] = 0.f;

    const unsigned short* kbase = f2t + (size_t)b * NHW * NC;
#pragma unroll 1
    for (int jb = 0; jb < NHW; jb += 32) {
        __builtin_prefetch(kbase + (size_t)(jb + 32 + nn) * NC, 0, 1);
        Acc slo, shi;
#pragma unroll
        for (int r = 0; r < 8; r++) { slo.f[r] = 0.f; shi.f[r] = 0.f; }
#pragma unroll
        for (int kk = 0; kk < 6; kk++) {
            Frag bl, bh;
            loadB(bl, kbase + (size_t)(jb + nn) * NC + kk * 32, g);
            loadB(bh, kbase + (size_t)(jb + 16 + nn) * NC + kk * 32, g);
            slo.v = wmma16(qf[kk], bl, slo.v);
            shi.v = wmma16(qf[kk], bh, shi.v);
        }
        // p = exp(s - m)/l ; C-layout -> A-layout transpose through LDS
#pragma unroll
        for (int r = 0; r < 8; r++) {
            int M = r + 8 * g;
            ptile[wv][M * 32 + nn]      = f2h(__expf(slo.f[r] - mrow[r]) * linv[r]);
            ptile[wv][M * 32 + 16 + nn] = f2h(__expf(shi.f[r] - mrow[r]) * linv[r]);
        }
        asm volatile("s_wait_dscnt 0" ::: "memory");   // in-wave cross-lane LDS visibility
        Frag pf;
        pf.q[0] = *(const u32x4*)&ptile[wv][nn * 32 + g * 8];
        pf.q[1] = *(const u32x4*)&ptile[wv][nn * 32 + 16 + g * 8];
#pragma unroll
        for (int cg = 0; cg < 12; cg++) {
            Frag bv;
            loadB(bv, vt + ((size_t)b * NC + cg * 16 + nn) * NHW + jb, g);
            acc[cg].v = wmma16(pf, bv, acc[cg].v);
        }
    }
    // epilogue with query-side pooling
    if (mode == 0) {
#pragma unroll
        for (int cg = 0; cg < 12; cg++)
#pragma unroll
            for (int r = 0; r < 8; r++) {
                int cc = cg * 16 + nn;
                int i = t * 16 + r + 8 * g;
                outp[((size_t)b * NC + cc) * NHW + i] = acc[cg].f[r];
            }
    } else if (mode == 1) {
        int y0 = (t >> 3) * 2, x0 = (t & 7) * 8;
        int qbase = (y0 >> 1) * 32 + (x0 >> 1);
#pragma unroll
        for (int cg = 0; cg < 12; cg++)
#pragma unroll
            for (int tt = 0; tt < 4; tt++) {
                float v = acc[cg].f[2 * tt] + acc[cg].f[2 * tt + 1];
                v += __shfl_xor(v, 16, 32);            // add the other y row
                if (g == 0)
                    outp[((size_t)b * NC + cg * 16 + nn) * 1024 + qbase + tt] = v * (1.f / 16.f);
            }
    } else {
        int y0 = (t >> 4) * 4, x0 = (t & 15) * 4;
        int qp = (y0 >> 2) * 16 + (x0 >> 2);
#pragma unroll
        for (int cg = 0; cg < 12; cg++) {
            float v = 0.f;
#pragma unroll
            for (int r = 0; r < 8; r++) v += acc[cg].f[r];
            v += __shfl_xor(v, 16, 32);
            if (g == 0)
                outp[((size_t)b * NC + cg * 16 + nn) * 256 + qp] = v * (1.f / 256.f);
        }
    }
}

// ------------------------------ launcher --------------------------------
extern "C" void kernel_launch(void* const* d_in, const int* in_sizes, int n_in,
                              void* d_out, int out_size, void* d_ws, size_t ws_size,
                              hipStream_t stream) {
    (void)in_sizes; (void)n_in; (void)out_size; (void)ws_size;
    const float* fd1  = (const float*)d_in[0];
    const float* fd2  = (const float*)d_in[1];
    const float* reff = (const float*)d_in[2];
    const float* o1   = (const float*)d_in[3];
    const float* o2   = (const float*)d_in[4];
    const float* tw1  = (const float*)d_in[5];
    const float* tb1  = (const float*)d_in[6];
    const float* tw2  = (const float*)d_in[7];
    const float* tb2  = (const float*)d_in[8];
    const float* pw1  = (const float*)d_in[9];
    const float* pb1  = (const float*)d_in[10];
    const float* pw2  = (const float*)d_in[11];
    const float* pb2  = (const float*)d_in[12];

    char* ws = (char*)d_ws;
    size_t off = 0;
    auto alloc = [&](size_t bytes) {
        void* p = ws + off; off += (bytes + 255) & ~(size_t)255; return p;
    };
    const size_t padB  = (size_t)NB * NPHW * NC * 2;  // padded HWC f16
    const size_t featB = (size_t)NB * NHW * NC * 2;   // plain f16 map
    const size_t wgtB  = (size_t)NC * NKW * 2;
    unsigned short* xh1 = (unsigned short*)alloc(padB);   // padded dec1
    unsigned short* xh2 = (unsigned short*)alloc(padB);   // padded dec2
    unsigned short* y1  = (unsigned short*)alloc(padB);   // padded conv1(theta)
    unsigned short* y2  = (unsigned short*)alloc(padB);   // padded conv1(phi)
    unsigned short* wh0 = (unsigned short*)alloc(wgtB);
    unsigned short* wh1 = (unsigned short*)alloc(wgtB);
    unsigned short* wh2 = (unsigned short*)alloc(wgtB);
    unsigned short* wh3 = (unsigned short*)alloc(wgtB);
    unsigned short* f1h = (unsigned short*)alloc(featB);  // [B,HW,C]
    unsigned short* f2h_= (unsigned short*)alloc(featB);  // [B,HW,C]
    unsigned short* v0t = (unsigned short*)alloc(featB);  // [B,C,HW]
    unsigned short* v1t = (unsigned short*)alloc(featB);
    unsigned short* v2t = (unsigned short*)alloc(featB);
    float* mbuf = (float*)alloc((size_t)NB * NHW * 4);
    float* lbuf = (float*)alloc((size_t)NB * NHW * 4);

    float* out0 = (float*)d_out;
    float* out1 = out0 + (size_t)NB * NC * NHW;
    float* out2 = out1 + (size_t)NB * NC * 1024;

    const int nPad  = NB * NPHW * NC;
    const int nFeat = NB * NC * NHW;
    const int nWgt  = NC * NKW;
    pack_pad_hwc<<<512, 256, 0, stream>>>(fd1, xh1, nPad);
    pack_pad_hwc<<<512, 256, 0, stream>>>(fd2, xh2, nPad);
    fill_zero_u16<<<512, 256, 0, stream>>>(y1, nPad);     // borders of conv1 outputs
    fill_zero_u16<<<512, 256, 0, stream>>>(y2, nPad);
    cvt_f16<<<512, 256, 0, stream>>>(reff, v0t, nFeat);   // CHW == [C, j] already
    pack_wgt<<<256, 256, 0, stream>>>(tw1, wh0, nWgt);
    pack_wgt<<<256, 256, 0, stream>>>(tw2, wh1, nWgt);
    pack_wgt<<<256, 256, 0, stream>>>(pw1, wh2, nWgt);
    pack_wgt<<<256, 256, 0, stream>>>(pw2, wh3, nWgt);
    upsample_f16<<<512, 256, 0, stream>>>(o1, v1t, 2, nFeat);
    upsample_f16<<<512, 256, 0, stream>>>(o2, v2t, 4, nFeat);

    // resblocks (theta on feature_dec1, phi on feature_dec2)
    conv16<<<3072, 128, 0, stream>>>(xh1, wh0, tb1, nullptr, y1, 0);
    conv16<<<3072, 128, 0, stream>>>(y1,  wh1, tb2, fd1,     f1h, 1);
    conv16<<<3072, 128, 0, stream>>>(xh2, wh2, pb1, nullptr, y2, 0);
    conv16<<<3072, 128, 0, stream>>>(y2,  wh3, pb2, fd2,     f2h_, 1);

    // fused non-local attention (flash-style; corr never materialized)
    attn_stats<<<256, 128, 0, stream>>>(f1h, f2h_, mbuf, lbuf);
    attn_out<<<256, 128, 0, stream>>>(f1h, f2h_, v0t, mbuf, lbuf, out0, 0);
    attn_out<<<256, 128, 0, stream>>>(f1h, f2h_, v1t, mbuf, lbuf, out1, 1);
    attn_out<<<256, 128, 0, stream>>>(f1h, f2h_, v2t, mbuf, lbuf, out2, 2);
}
